// TransformerBlockQuantum_65481071395350
// MI455X (gfx1250) — matrix operand order repaired and checked by reading, not compiled, not run on hardware
//
#include <hip/hip_runtime.h>
#include <hip/hip_bf16.h>

// The Python harness owns main(), I/O, device malloc/memcpy, and timing.
// We implement kernel_launch() + kernels only. All launches on `stream`.

typedef __attribute__((ext_vector_type(16))) _Float16 v16h;
typedef __attribute__((ext_vector_type(8)))  float    v8f;

#define E_DIM   768
#define FFN_DIM 3072
#define NQ      8
#define DKQ     64
#define BM      32           // rows per workgroup (2 M-tiles per wave)
#define NMT     2            // M-tiles per wave
#define NWAVES  8
#define NT_PER_WAVE 6        // 48 n-tiles / 8 waves
#define KT1     (E_DIM/32)   // 24
#define KT2     (FFN_DIM/32) // 96
#define LN_EPS  1e-5f
#define ROWS_PER_WAVE (BM/NWAVES)   // 4 rows per wave for layernorms

#define HS_STRIDE (FFN_DIM + 8)   // pad to kill LDS bank conflicts on A gathers
#define AS_STRIDE 40

// dynamic shared memory layout (bytes)
#define XS_OFF   0
#define XS_BYTES (BM*E_DIM*4)             // 98304: x rows, reused as x1
#define HS_OFF   (XS_OFF + XS_BYTES)
#define HS_BYTES (BM*HS_STRIDE*2)         // 197120: h (f16), aliased as ffn f32 staging
#define AS_OFF   (HS_OFF + HS_BYTES)
#define AS_BYTES (2*BM*AS_STRIDE*2)       // 5120: double-buffered cos(x+rx) A staging
#define QS_OFF   (AS_OFF + AS_BYTES)
#define QS_BYTES (BM*NQ*4)                // 1024
#define RX_OFF   (QS_OFF + QS_BYTES)
#define RX_BYTES (DKQ*4)
#define SMEM_BYTES (RX_OFF + RX_BYTES)    // ~294.8 KB (< 320 KB WGP LDS)

// ---------------------------------------------------------------------------
// Swizzle a (K,N) fp32 weight matrix into WMMA B-fragment-major f16 tiles.
// Tile (tk,tn) -> 512 halves at offset (tn*KT + tk)*512; within a tile,
// lane-major: lane L (n=L&15, hi=L>>4) holds halves [2j,2j+1] = B[tk*32 +
// 2*(j&3) + 16*(j>>2) + 8*hi][tn*16 + n]  (ISA 7.12.2: 32x16 16-bit B layout).
// ---------------------------------------------------------------------------
__global__ void qtb_swizzle_b(const float* __restrict__ W,
                              _Float16* __restrict__ outp, int K, int N) {
    int KT = K >> 5;
    int total = KT * (N >> 4) * 32;
    int gid = blockIdx.x * blockDim.x + threadIdx.x;
    if (gid >= total) return;
    int lane = gid & 31;
    int tile = gid >> 5;
    int tk = tile % KT;
    int tn = tile / KT;
    int n  = lane & 15;
    int hi = lane >> 4;
    _Float16* dst = outp + (size_t)tile * 512 + lane * 16;
    int col = tn * 16 + n;
#pragma unroll
    for (int j = 0; j < 8; ++j) {
        int k = tk * 32 + (j & 3) * 2 + (j >> 2) * 16 + hi * 8;
        dst[2 * j]     = (_Float16)W[(size_t)k       * N + col];
        dst[2 * j + 1] = (_Float16)W[(size_t)(k + 1) * N + col];
    }
}

// ---------------------------------------------------------------------------
// Fully fused quantum transformer block, one 32-row block per workgroup.
// Each wave: 2 M-tiles x 6 N-tiles -> every B fragment feeds 2 WMMAs.
// ---------------------------------------------------------------------------
__global__ __launch_bounds__(256)
void qtb_fused(const float* __restrict__ x,  const float* __restrict__ rx,
               const float* __restrict__ ry,
               const _Float16* __restrict__ WcS, const float* __restrict__ bc,
               const float* __restrict__ W1, const float* __restrict__ b1,
               const _Float16* __restrict__ W2S, const float* __restrict__ b2,
               const float* __restrict__ g1, const float* __restrict__ be1,
               const float* __restrict__ g2, const float* __restrict__ be2,
               float* __restrict__ out) {
    extern __shared__ unsigned char smem[];
    float*    xs  = (float*)(smem + XS_OFF);
    _Float16* hs  = (_Float16*)(smem + HS_OFF);
    float*    fs  = (float*)(smem + HS_OFF);     // alias: ffn f32 staging
    _Float16* as  = (_Float16*)(smem + AS_OFF);  // 2 buffers of BM*AS_STRIDE
    float*    qs  = (float*)(smem + QS_OFF);
    float*    rxs = (float*)(smem + RX_OFF);

    const int tid  = threadIdx.x;
    const int lane = tid & 31;
    const int wv   = tid >> 5;
    const int r0   = blockIdx.x * BM;
    const int n    = lane & 15;
    const int hi   = lane >> 4;

    if (tid < DKQ) rxs[tid] = rx[tid];
    for (int i = tid; i < BM * E_DIM; i += 256) {
        int m = i / E_DIM, e = i - m * E_DIM;
        xs[m * E_DIM + e] = x[(size_t)(r0 + m) * E_DIM + e];
    }

    // ---------------- GEMM1: attn = cos(x+rx) @ Wc ----------------
    v8f c[NMT][NT_PER_WAVE];
#pragma unroll
    for (int mt = 0; mt < NMT; ++mt)
#pragma unroll
        for (int t = 0; t < NT_PER_WAVE; ++t)
#pragma unroll
            for (int r = 0; r < 8; ++r) c[mt][t][r] = 0.f;

    __syncthreads();
    // stage K-tile 0 (BM x 32 cos values)
    for (int i = tid; i < BM * 32; i += 256) {
        int m = i >> 5, j = i & 31;
        as[m * AS_STRIDE + j] = (_Float16)__cosf(xs[m * E_DIM + j] + rxs[j & (DKQ - 1)]);
    }
    __syncthreads();

    const v16h* WcV = (const v16h*)WcS;
    for (int kk = 0; kk < KT1; ++kk) {
        const int cur = kk & 1;
        // stage next K-tile into other buffer (overlaps with WMMAs below)
        if (kk + 1 < KT1) {
            int kbase = (kk + 1) * 32;
            _Float16* asn = as + (cur ^ 1) * BM * AS_STRIDE;
            for (int i = tid; i < BM * 32; i += 256) {
                int m = i >> 5, j = i & 31;
                int k = kbase + j;
                asn[m * AS_STRIDE + j] =
                    (_Float16)__cosf(xs[m * E_DIM + k] + rxs[k & (DKQ - 1)]);
            }
        }
        // A fragments for the 2 M-tiles (ISA 16x32 f16 A layout; contiguous 16B chunks)
        const _Float16* asc = as + cur * BM * AS_STRIDE;
        v16h a[NMT];
#pragma unroll
        for (int mt = 0; mt < NMT; ++mt)
#pragma unroll
            for (int j = 0; j < 8; ++j) {
                int kb = ((j & 3) * 2 + (j >> 2) * 16) + hi * 8;
                a[mt][2 * j]     = asc[(mt * 16 + n) * AS_STRIDE + kb];
                a[mt][2 * j + 1] = asc[(mt * 16 + n) * AS_STRIDE + kb + 1];
            }
#pragma unroll
        for (int t = 0; t < NT_PER_WAVE; ++t) {
            int tn = wv * NT_PER_WAVE + t;
            v16h b = WcV[(size_t)(tn * KT1 + kk) * 32 + lane];
            c[0][t] = __builtin_amdgcn_wmma_f32_16x16x32_f16(
                          false, a[0], false, b, (short)0, c[0][t], false, false);
            c[1][t] = __builtin_amdgcn_wmma_f32_16x16x32_f16(
                          false, a[1], false, b, (short)0, c[1][t], false, false);
        }
        __syncthreads();   // one barrier per K-step (double-buffered staging)
    }
    // residual add + bias into xs
#pragma unroll
    for (int t = 0; t < NT_PER_WAVE; ++t) {
        int col = (wv * NT_PER_WAVE + t) * 16 + n;
        float bcv = bc[col];
#pragma unroll
        for (int mt = 0; mt < NMT; ++mt)
#pragma unroll
            for (int r = 0; r < 8; ++r) {
                int row = mt * 16 + r + hi * 8;
                xs[row * E_DIM + col] += c[mt][t][r] + bcv;
            }
    }
    __syncthreads();

    // ---------------- LN1 in place (xs -> x1) ----------------
    for (int m = wv * ROWS_PER_WAVE; m < wv * ROWS_PER_WAVE + ROWS_PER_WAVE; ++m) {
        float s1 = 0.f, s2 = 0.f;
        for (int e = lane; e < E_DIM; e += 32) {
            float v = xs[m * E_DIM + e]; s1 += v; s2 += v * v;
        }
#pragma unroll
        for (int o = 16; o > 0; o >>= 1) {
            s1 += __shfl_xor(s1, o, 32); s2 += __shfl_xor(s2, o, 32);
        }
        float mean = s1 * (1.f / E_DIM);
        float var  = s2 * (1.f / E_DIM) - mean * mean;
        float rstd = rsqrtf(var + LN_EPS);
        for (int e = lane; e < E_DIM; e += 32) {
            float v = xs[m * E_DIM + e];
            xs[m * E_DIM + e] = (v - mean) * rstd * g1[e] + be1[e];
        }
    }
    __syncthreads();

    // ---------------- quantum FFN front: q, h ----------------
    {   // BM*NQ == 256 == blockDim
        int m = tid >> 3, i = tid & 7;
        qs[m * NQ + i] = __cosf(xs[m * E_DIM + i]) * __cosf(ry[i]);
    }
    __syncthreads();
    for (int m = 0; m < BM; ++m) {
        for (int f = tid; f < FFN_DIM; f += 256) {
            float acc = b1[f];
#pragma unroll
            for (int i = 0; i < NQ; ++i) acc += qs[m * NQ + i] * W1[i * FFN_DIM + f];
            hs[m * HS_STRIDE + f] = (_Float16)fmaxf(acc, 0.f);
        }
    }
    __syncthreads();

    // ---------------- GEMM3: ffn = h @ W2 ----------------
    v8f d[NMT][NT_PER_WAVE];
#pragma unroll
    for (int mt = 0; mt < NMT; ++mt)
#pragma unroll
        for (int t = 0; t < NT_PER_WAVE; ++t)
#pragma unroll
            for (int r = 0; r < 8; ++r) d[mt][t][r] = 0.f;

    const v16h* W2V = (const v16h*)W2S;
    for (int kk = 0; kk < KT2; ++kk) {
        v16h a[NMT];
#pragma unroll
        for (int mt = 0; mt < NMT; ++mt)
#pragma unroll
            for (int j = 0; j < 8; ++j) {
                int kb = kk * 32 + ((j & 3) * 2 + (j >> 2) * 16) + hi * 8;
                a[mt][2 * j]     = hs[(mt * 16 + n) * HS_STRIDE + kb];
                a[mt][2 * j + 1] = hs[(mt * 16 + n) * HS_STRIDE + kb + 1];
            }
#pragma unroll
        for (int t = 0; t < NT_PER_WAVE; ++t) {
            int tn = wv * NT_PER_WAVE + t;
            const v16h* bp = W2V + (size_t)(tn * KT2 + kk) * 32 + lane;
            if (kk + 1 < KT2)
                __builtin_prefetch((const void*)(bp + 32), 0, 0);  // next K tile
            v16h b = *bp;
            d[0][t] = __builtin_amdgcn_wmma_f32_16x16x32_f16(
                          false, a[0], false, b, (short)0, d[0][t], false, false);
            d[1][t] = __builtin_amdgcn_wmma_f32_16x16x32_f16(
                          false, a[1], false, b, (short)0, d[1][t], false, false);
        }
    }
    __syncthreads();                       // hs reads done; safe to alias as fs
#pragma unroll
    for (int t = 0; t < NT_PER_WAVE; ++t) {
        int col = (wv * NT_PER_WAVE + t) * 16 + n;
        float b2v = b2[col];
#pragma unroll
        for (int mt = 0; mt < NMT; ++mt)
#pragma unroll
            for (int r = 0; r < 8; ++r) {
                int row = mt * 16 + r + hi * 8;
                fs[row * E_DIM + col] = d[mt][t][r] + b2v;
            }
    }
    __syncthreads();

    // ---------------- LN2 + output ----------------
    for (int m = wv * ROWS_PER_WAVE; m < wv * ROWS_PER_WAVE + ROWS_PER_WAVE; ++m) {
        float s1 = 0.f, s2 = 0.f;
        for (int e = lane; e < E_DIM; e += 32) {
            float v = xs[m * E_DIM + e] + fs[m * E_DIM + e];
            s1 += v; s2 += v * v;
        }
#pragma unroll
        for (int o = 16; o > 0; o >>= 1) {
            s1 += __shfl_xor(s1, o, 32); s2 += __shfl_xor(s2, o, 32);
        }
        float mean = s1 * (1.f / E_DIM);
        float var  = s2 * (1.f / E_DIM) - mean * mean;
        float rstd = rsqrtf(var + LN_EPS);
        size_t rg = (size_t)(r0 + m) * E_DIM;
        for (int e = lane; e < E_DIM; e += 32) {
            float v = xs[m * E_DIM + e] + fs[m * E_DIM + e];
            out[rg + e] = (v - mean) * rstd * g2[e] + be2[e];
        }
    }
}

extern "C" void kernel_launch(void* const* d_in, const int* in_sizes, int n_in,
                              void* d_out, int out_size, void* d_ws, size_t ws_size,
                              hipStream_t stream) {
    const float* x   = (const float*)d_in[0];
    const float* rx  = (const float*)d_in[1];
    const float* ry  = (const float*)d_in[2];
    const float* Wc  = (const float*)d_in[3];
    const float* bc  = (const float*)d_in[4];
    const float* W1  = (const float*)d_in[5];
    const float* b1  = (const float*)d_in[6];
    const float* W2  = (const float*)d_in[7];
    const float* b2  = (const float*)d_in[8];
    const float* g1  = (const float*)d_in[9];
    const float* be1 = (const float*)d_in[10];
    const float* g2  = (const float*)d_in[11];
    const float* be2 = (const float*)d_in[12];
    float* out = (float*)d_out;

    // workspace: swizzled f16 Wc (1.125 MB) then swizzled f16 W2 (4.5 MB)
    _Float16* WcS = (_Float16*)d_ws;
    _Float16* W2S = (_Float16*)((char*)d_ws + (size_t)E_DIM * E_DIM * sizeof(_Float16));

    {   // Wc: 768x768 -> 24*48 tiles * 32 lanes
        int threads = KT1 * (E_DIM / 16) * 32;
        qtb_swizzle_b<<<(threads + 255) / 256, 256, 0, stream>>>(Wc, WcS, E_DIM, E_DIM);
    }
    {   // W2: 3072x768 -> 96*48 tiles * 32 lanes
        int threads = KT2 * (E_DIM / 16) * 32;
        qtb_swizzle_b<<<(threads + 255) / 256, 256, 0, stream>>>(W2, W2S, FFN_DIM, E_DIM);
    }

    (void)hipFuncSetAttribute((const void*)qtb_fused,
                              hipFuncAttributeMaxDynamicSharedMemorySize,
                              (int)SMEM_BYTES);

    int mblocks = (8 * 2048) / BM;   // 512 workgroups of 32 rows
    qtb_fused<<<mblocks, 256, SMEM_BYTES, stream>>>(
        x, rx, ry, WcS, bc, W1, b1, W2S, b2, g1, be1, g2, be2, out);
}
